// TransformerBlock_38852274160164
// MI455X (gfx1250) — matrix operand order repaired and checked by reading, not compiled, not run on hardware
//
#include <hip/hip_runtime.h>
#include <math.h>

typedef __bf16 bf16;
typedef __attribute__((ext_vector_type(16))) __bf16 v16bf;
typedef __attribute__((ext_vector_type(8)))  __bf16 v8bf;
typedef __attribute__((ext_vector_type(8)))  float  v8f;

#define B_   8
#define S_   1024
#define D_   768
#define H_   12
#define HS_  64
#define F_   3072
#define M_   (B_*S_)   // 8192 rows

// ---------------------------------------------------------------------------
// WMMA helpers (CDNA5 wave32, 16x16x32 bf16 -> f32)
// ---------------------------------------------------------------------------
__device__ __forceinline__ v8f wmma_bf16(v16bf a, v16bf b, v8f c) {
  return __builtin_amdgcn_wmma_f32_16x16x32_bf16(false, a, false, b, (short)0, c,
                                                 false, false);
}

// A-matrix 16x32 bf16 fragment. ISA layout:
// lanes 0-15 : row = lane,    K in {0..7, 16..23}
// lanes 16-31: row = lane-16, K in {8..15, 24..31}
__device__ __forceinline__ v16bf load_a_frag(const bf16* base, int lda, int k0, int lane) {
  const int half = lane >> 4;
  const int m    = lane & 15;
  const bf16* p  = base + (size_t)m * lda + k0 + half * 8;
  v8bf lo = *(const v8bf*)(p);        // K = k0+half*8 .. +7
  v8bf hi = *(const v8bf*)(p + 16);   // K = k0+half*8+16 .. +23
  return __builtin_shufflevector(lo, hi, 0,1,2,3,4,5,6,7,8,9,10,11,12,13,14,15);
}

// B-matrix 32x16 bf16 fragment from "Bt" storage: Bt[n][k] (column-major B).
// lanes 0-15 : col n = n0+lane,    K in {0..7, 16..23}
// lanes 16-31: col n = n0+lane-16, K in {8..15, 24..31}
__device__ __forceinline__ v16bf load_b_frag(const bf16* base, int ldb, int n0, int k0, int lane) {
  const int n   = n0 + (lane & 15);
  const bf16* p = base + (size_t)n * ldb + k0 + (lane >> 4) * 8;
  v8bf lo = *(const v8bf*)(p);
  v8bf hi = *(const v8bf*)(p + 16);
  return __builtin_shufflevector(lo, hi, 0,1,2,3,4,5,6,7,8,9,10,11,12,13,14,15);
}

__device__ __forceinline__ float gelu_tanh(float x) {
  const float c = 0.7978845608028654f;   // sqrt(2/pi)
  float x3 = x * x * x;
  return 0.5f * x * (1.f + tanhf(c * (x + 0.044715f * x3)));
}

// ---------------------------------------------------------------------------
// Elementwise conversion kernels
// ---------------------------------------------------------------------------
__global__ __launch_bounds__(256) void cvt_bf16(const float* __restrict__ a,
                                                bf16* __restrict__ o, int n) {
  int i = blockIdx.x * 256 + threadIdx.x;
  if (i < n) o[i] = (bf16)a[i];
}

// W[K][N] fp32 -> Wt[N][K] bf16
__global__ __launch_bounds__(256) void wtrans(const float* __restrict__ W,
                                              bf16* __restrict__ Wt, int K, int N) {
  int i = blockIdx.x * 256 + threadIdx.x;
  if (i >= K * N) return;
  int n = i / K;
  int k = i - n * K;
  Wt[i] = (bf16)W[(size_t)k * N + n];
}

// ---------------------------------------------------------------------------
// Generic WMMA GEMM:  out[M][N] = A[M][K] * Bt[N][K]^T + bias
// One wave computes a 32x64 tile (2 row-subtiles x 4 col-subtiles): each
// B fragment feeds 2 WMMAs, each A fragment feeds 4 -> 1.5 b128 loads/WMMA.
// EPI: 0=f32 store, 1=bf16, 2=bf16*scale, 3=gelu->bf16,
//      4=bf16 transposed V store (vT[b][h][d][s]).
// ---------------------------------------------------------------------------
template<int EPI>
__global__ __launch_bounds__(256) void gemm_wmma(const bf16* __restrict__ A,
                                                 const bf16* __restrict__ Bt,
                                                 const float* __restrict__ bias,
                                                 void* __restrict__ outp,
                                                 int M, int N, int K, float scale) {
  const int lane = threadIdx.x & 31;
  const int gw   = blockIdx.x * 8 + (threadIdx.x >> 5);
  const int tiles_n = N >> 6;
  const int tm = gw / tiles_n;
  const int tn = gw - tm * tiles_n;
  if (tm * 32 >= M) return;                 // wave-uniform guard
  const int m0 = tm * 32, n0 = tn * 64;

  v8f acc[2][4] = {};
  const bf16* arow0 = A + (size_t)m0 * K;
  const bf16* arow1 = A + (size_t)(m0 + 16) * K;
  for (int k0 = 0; k0 < K; k0 += 32) {
    v16bf a0 = load_a_frag(arow0, K, k0, lane);
    v16bf a1 = load_a_frag(arow1, K, k0, lane);
#pragma unroll
    for (int j = 0; j < 4; ++j) {
      v16bf b = load_b_frag(Bt, K, n0 + j * 16, k0, lane);
      acc[0][j] = wmma_bf16(a0, b, acc[0][j]);
      acc[1][j] = wmma_bf16(a1, b, acc[1][j]);
    }
  }

  const int half = lane >> 4, nc = lane & 15;
#pragma unroll
  for (int i = 0; i < 2; ++i) {
#pragma unroll
    for (int j = 0; j < 4; ++j) {
#pragma unroll
      for (int r = 0; r < 8; ++r) {
        int m = m0 + i * 16 + r + half * 8;
        int n = n0 + j * 16 + nc;
        float v = acc[i][j][r] + bias[n];
        if (EPI == 2) v = v * scale;
        if (EPI == 3) v = gelu_tanh(v);
        if (EPI == 0) {
          ((float*)outp)[(size_t)m * N + n] = v;
        } else if (EPI == 4) {
          int bb = m >> 10, s = m & (S_ - 1);
          int h = n >> 6,  d = n & (HS_ - 1);
          ((bf16*)outp)[(((size_t)bb * H_ + h) * HS_ + d) * S_ + s] = (bf16)v;
        } else {
          ((bf16*)outp)[(size_t)m * N + n] = (bf16)v;
        }
      }
    }
  }
}

// ---------------------------------------------------------------------------
// Flash-style attention. One wave = one (b, h, 16-row q-tile).
// Q pre-scaled by 1/sqrt(64). K stored row-major [b*S][D] (cols h*64..),
// V stored transposed vT[b][h][d][s].
// ---------------------------------------------------------------------------
__global__ __launch_bounds__(256) void attn_wmma(const bf16* __restrict__ Q,
                                                 const bf16* __restrict__ Kb,
                                                 const bf16* __restrict__ Vt,
                                                 const int*  __restrict__ mask,
                                                 bf16* __restrict__ ctx) {
  __shared__ __align__(16) bf16 plds[8 * 512];   // per-wave 16x32 P tile
  const int lane = threadIdx.x & 31;
  const int wv   = threadIdx.x >> 5;
  const int gw   = blockIdx.x * 8 + wv;
  const int b    = gw / (H_ * (S_ / 16));
  int rem        = gw - b * (H_ * (S_ / 16));
  const int h    = rem / (S_ / 16);
  const int qt   = rem - h * (S_ / 16);
  const int q0   = qt * 16;
  const int half = lane >> 4, nc = lane & 15;

  const bf16* qbase = Q  + ((size_t)b * S_ + q0) * D_ + h * HS_;
  const bf16* kbase = Kb + (size_t)b * S_ * D_ + h * HS_;
  const bf16* vbase = Vt + ((size_t)b * H_ + h) * HS_ * S_;
  const int*  mbase = mask + ((size_t)b * S_ + q0) * S_;   // mask[b][0][q][k]

  v16bf qa0 = load_a_frag(qbase, D_, 0,  lane);
  v16bf qa1 = load_a_frag(qbase, D_, 32, lane);

  v8f acc[4] = {};
  float mr[8], lr[8];
#pragma unroll
  for (int r = 0; r < 8; ++r) { mr[r] = -1e30f; lr[r] = 0.f; }

  bf16* myl = plds + wv * 512;

  for (int kt = 0; kt < S_ / 32; ++kt) {
    const int kk0 = kt * 32;
    // S = Q * K^T  (two 16-key subtiles, reduction over d=64)
    v8f s0 = {}, s1 = {};
    s0 = wmma_bf16(qa0, load_b_frag(kbase, D_, kk0,      0,  lane), s0);
    s0 = wmma_bf16(qa1, load_b_frag(kbase, D_, kk0,      32, lane), s0);
    s1 = wmma_bf16(qa0, load_b_frag(kbase, D_, kk0 + 16, 0,  lane), s1);
    s1 = wmma_bf16(qa1, load_b_frag(kbase, D_, kk0 + 16, 32, lane), s1);

    // mask
#pragma unroll
    for (int r = 0; r < 8; ++r) {
      int q = r + half * 8;
      const int* mp = mbase + (size_t)q * S_ + kk0 + nc;
      if (mp[0]  == 0) s0[r] = -1e9f;
      if (mp[16] == 0) s1[r] = -1e9f;
    }

    // online softmax: row lives across 16 lanes of a half-wave
    float p0[8], p1[8], sc[8];
#pragma unroll
    for (int r = 0; r < 8; ++r) {
      float v = fmaxf(s0[r], s1[r]);
#pragma unroll
      for (int msk = 1; msk < 16; msk <<= 1) v = fmaxf(v, __shfl_xor(v, msk, 32));
      float mnew = fmaxf(mr[r], v);
      sc[r] = __expf(mr[r] - mnew);
      mr[r] = mnew;
      p0[r] = __expf(s0[r] - mnew);
      p1[r] = __expf(s1[r] - mnew);
      float t = p0[r] + p1[r];
#pragma unroll
      for (int msk = 1; msk < 16; msk <<= 1) t += __shfl_xor(t, msk, 32);
      lr[r] = lr[r] * sc[r] + t;
    }
#pragma unroll
    for (int j = 0; j < 4; ++j)
#pragma unroll
      for (int r = 0; r < 8; ++r) acc[j][r] = acc[j][r] * sc[r];

    // stage P (C-layout) through LDS to rebuild the A-fragment layout
#pragma unroll
    for (int r = 0; r < 8; ++r) {
      int row = r + half * 8;
      myl[row * 32 + nc]      = (bf16)p0[r];
      myl[row * 32 + 16 + nc] = (bf16)p1[r];
    }
    __syncthreads();
    const bf16* pp = myl + (size_t)(lane & 15) * 32 + half * 8;
    v8bf lo = *(const v8bf*)pp;
    v8bf hi = *(const v8bf*)(pp + 16);
    v16bf pa = __builtin_shufflevector(lo, hi, 0,1,2,3,4,5,6,7,8,9,10,11,12,13,14,15);
    __syncthreads();

    // ctx += P * V   (B-frag contiguous thanks to vT layout)
#pragma unroll
    for (int j = 0; j < 4; ++j)
      acc[j] = wmma_bf16(pa, load_b_frag(vbase, S_, j * 16, kk0, lane), acc[j]);
  }

  float inv[8];
#pragma unroll
  for (int r = 0; r < 8; ++r) inv[r] = 1.f / lr[r];
#pragma unroll
  for (int j = 0; j < 4; ++j)
#pragma unroll
    for (int r = 0; r < 8; ++r) {
      int m   = q0 + r + half * 8;
      int col = h * HS_ + j * 16 + nc;
      ctx[((size_t)b * S_ + m) * D_ + col] = (bf16)(acc[j][r] * inv[r]);
    }
}

// ---------------------------------------------------------------------------
// Fused residual + LayerNorm (torch-style: ddof=1, eps added to std).
// One block (256 threads) per row of 768. Emits f32 and optional bf16.
// ---------------------------------------------------------------------------
__global__ __launch_bounds__(256) void ln_fused(const float* __restrict__ xin,
                                                const float* __restrict__ yin,
                                                const float* __restrict__ w,
                                                const float* __restrict__ bb,
                                                float* __restrict__ outf,
                                                bf16* __restrict__ outb) {
  __shared__ float red[16];
  const int row = blockIdx.x;
  const int tid = threadIdx.x;
  const float* px = xin + (size_t)row * D_;
  const float* py = yin + (size_t)row * D_;
  float v[3], s = 0.f, ss = 0.f;
#pragma unroll
  for (int i = 0; i < 3; ++i) {
    int c = tid + i * 256;
    v[i] = px[c] + py[c];
    s += v[i]; ss += v[i] * v[i];
  }
#pragma unroll
  for (int m = 1; m < 32; m <<= 1) { s += __shfl_xor(s, m, 32); ss += __shfl_xor(ss, m, 32); }
  if ((tid & 31) == 0) { red[tid >> 5] = s; red[8 + (tid >> 5)] = ss; }
  __syncthreads();
  s = 0.f; ss = 0.f;
#pragma unroll
  for (int k = 0; k < 8; ++k) { s += red[k]; ss += red[8 + k]; }
  float mean = s * (1.f / D_);
  float var  = (ss - (float)D_ * mean * mean) * (1.f / (D_ - 1));
  var = fmaxf(var, 0.f);
  float rstd = 1.f / (sqrtf(var) + 1e-6f);
#pragma unroll
  for (int i = 0; i < 3; ++i) {
    int c = tid + i * 256;
    float o = w[c] * (v[i] - mean) * rstd + bb[c];
    outf[(size_t)row * D_ + c] = o;
    if (outb) outb[(size_t)row * D_ + c] = (bf16)o;
  }
}

// ---------------------------------------------------------------------------
// Host-side orchestration
// ---------------------------------------------------------------------------
extern "C" void kernel_launch(void* const* d_in, const int* in_sizes, int n_in,
                              void* d_out, int out_size, void* d_ws, size_t ws_size,
                              hipStream_t stream) {
  (void)in_sizes; (void)n_in; (void)out_size; (void)ws_size;
  const float* x    = (const float*)d_in[0];
  const int*   mask = (const int*)d_in[1];
  const float* wq   = (const float*)d_in[2];
  const float* bq   = (const float*)d_in[3];
  const float* wk   = (const float*)d_in[4];
  const float* bk   = (const float*)d_in[5];
  const float* wvp  = (const float*)d_in[6];
  const float* bv   = (const float*)d_in[7];
  const float* wo   = (const float*)d_in[8];
  const float* bo   = (const float*)d_in[9];
  const float* w1   = (const float*)d_in[10];
  const float* b1   = (const float*)d_in[11];
  const float* w2   = (const float*)d_in[12];
  const float* b2   = (const float*)d_in[13];
  const float* ln1w = (const float*)d_in[14];
  const float* ln1b = (const float*)d_in[15];
  const float* ln2w = (const float*)d_in[16];
  const float* ln2b = (const float*)d_in[17];

  char* ws = (char*)d_ws;
  size_t off = 0;
  auto alloc = [&](size_t bytes) -> void* {
    void* p = ws + off;
    off += (bytes + 255) & ~(size_t)255;
    return p;
  };

  bf16* xb  = (bf16*)alloc((size_t)M_ * D_ * 2);
  bf16* wqT = (bf16*)alloc((size_t)D_ * D_ * 2);
  bf16* wkT = (bf16*)alloc((size_t)D_ * D_ * 2);
  bf16* wvT = (bf16*)alloc((size_t)D_ * D_ * 2);
  bf16* woT = (bf16*)alloc((size_t)D_ * D_ * 2);
  bf16* w1T = (bf16*)alloc((size_t)F_ * D_ * 2);
  bf16* w2T = (bf16*)alloc((size_t)D_ * F_ * 2);
  bf16* qb  = (bf16*)alloc((size_t)M_ * D_ * 2);
  bf16* kb  = (bf16*)alloc((size_t)M_ * D_ * 2);
  bf16* vT  = (bf16*)alloc((size_t)M_ * D_ * 2);
  bf16* cx  = (bf16*)alloc((size_t)M_ * D_ * 2);
  float* attn = (float*)alloc((size_t)M_ * D_ * 4);
  float* x1f  = (float*)alloc((size_t)M_ * D_ * 4);
  bf16*  x1b  = (bf16*)alloc((size_t)M_ * D_ * 2);
  bf16*  hb   = qb;     // FFN hidden [M, F] aliases dead q/k/vT/cx (exactly 48 MiB)
  float* y2   = attn;   // fc2 output aliases dead attn buffer

  dim3 blk(256);

  cvt_bf16<<<(M_ * D_ + 255) / 256, blk, 0, stream>>>(x, xb, M_ * D_);
  wtrans<<<(D_ * D_ + 255) / 256, blk, 0, stream>>>(wq,  wqT, D_, D_);
  wtrans<<<(D_ * D_ + 255) / 256, blk, 0, stream>>>(wk,  wkT, D_, D_);
  wtrans<<<(D_ * D_ + 255) / 256, blk, 0, stream>>>(wvp, wvT, D_, D_);
  wtrans<<<(D_ * D_ + 255) / 256, blk, 0, stream>>>(wo,  woT, D_, D_);
  wtrans<<<(D_ * F_ + 255) / 256, blk, 0, stream>>>(w1,  w1T, D_, F_);
  wtrans<<<(F_ * D_ + 255) / 256, blk, 0, stream>>>(w2,  w2T, F_, D_);

  // 32x64 tile per wave, 8 waves per block
  const int blocks_d = (M_ / 32) * (D_ / 64) / 8;   // 384
  const int blocks_f = (M_ / 32) * (F_ / 64) / 8;   // 1536

  // Q/K/V projections (Q scaled by 1/sqrt(head_size))
  gemm_wmma<2><<<blocks_d, blk, 0, stream>>>(xb, wqT, bq, qb, M_, D_, D_, 0.125f);
  gemm_wmma<1><<<blocks_d, blk, 0, stream>>>(xb, wkT, bk, kb, M_, D_, D_, 1.f);
  gemm_wmma<4><<<blocks_d, blk, 0, stream>>>(xb, wvT, bv, vT, M_, D_, D_, 1.f);

  attn_wmma<<<(B_ * H_ * (S_ / 16)) / 8, blk, 0, stream>>>(qb, kb, vT, mask, cx);

  gemm_wmma<0><<<blocks_d, blk, 0, stream>>>(cx, woT, bo, attn, M_, D_, D_, 1.f);
  ln_fused<<<M_, blk, 0, stream>>>(x, attn, ln1w, ln1b, x1f, x1b);

  gemm_wmma<3><<<blocks_f, blk, 0, stream>>>(x1b, w1T, b1, hb, M_, F_, D_, 1.f);
  gemm_wmma<0><<<blocks_d, blk, 0, stream>>>(hb, w2T, b2, y2, M_, D_, F_, 1.f);
  ln_fused<<<M_, blk, 0, stream>>>(x1f, y2, ln2w, ln2b, (float*)d_out, (bf16*)nullptr);
}
